// Attention_36146444763783
// MI455X (gfx1250) — compile-verified, matrix operand order checked by reading, not run on hardware
//
#include <hip/hip_runtime.h>

typedef _Float16 half_t;
typedef __attribute__((ext_vector_type(16))) _Float16 v16h;
typedef __attribute__((ext_vector_type(8)))  _Float16 v8h;
typedef __attribute__((ext_vector_type(4)))  _Float16 v4h;
typedef __attribute__((ext_vector_type(8)))  float    v8f;
typedef int vi4 __attribute__((vector_size(16)));   // matches builtin param type

#define CH      512
#define HEADS   8
#define HEAD_CH 64
#define HIDDEN  512
#define NQKV    1536
#define GROUPS  32
#define BATCH   16
#define NTOK    1024
#define ROWS    (BATCH * NTOK)   // 16384
#define EPS     1e-5f

#if __has_builtin(__builtin_amdgcn_global_load_async_to_lds_b128)
#define ASYNC_LDS 1
#endif

// ---------------------------------------------------------------------------
// WMMA fragment helpers (CDNA5 16-bit A/B layout, ISA 7.12.2):
//   lane m/n = lane & 15 ; lane-half selects K bank (+8)
//   fragment from a K-contiguous row = two 16-byte loads.
// ---------------------------------------------------------------------------
static __device__ __forceinline__ v16h frag_from(const half_t* row, int k0, int kb) {
  v8h lo = *(const v8h*)(row + k0 + kb);
  v8h hi = *(const v8h*)(row + k0 + kb + 16);
  v16h r;
#pragma unroll
  for (int i = 0; i < 8; ++i) { r[i] = lo[i]; r[i + 8] = hi[i]; }
  return r;
}

static __device__ __forceinline__ v8f wmma_f16(v16h a, v16h b, v8f c) {
  return __builtin_amdgcn_wmma_f32_16x16x32_f16(false, a, false, b, (short)0, c,
                                                false, false);
}

static __device__ __forceinline__ float redmax16(float v) {
#pragma unroll
  for (int m = 1; m < 16; m <<= 1) v = fmaxf(v, __shfl_xor(v, m, 32));
  return v;
}
static __device__ __forceinline__ float redsum16(float v) {
#pragma unroll
  for (int m = 1; m < 16; m <<= 1) v += __shfl_xor(v, m, 32);
  return v;
}

// 16-byte global -> LDS copy: async DMA path on CDNA5, sync fallback
static __device__ __forceinline__ void cp16(half_t* dst_lds, const half_t* src_g) {
#ifdef ASYNC_LDS
  __builtin_amdgcn_global_load_async_to_lds_b128(
      (__attribute__((address_space(1))) vi4*)src_g,
      (__attribute__((address_space(3))) vi4*)dst_lds, 0, 0);
#else
  *(v8h*)dst_lds = *(const v8h*)src_g;
#endif
}

template <int N>
static __device__ __forceinline__ void wait_async() {
#ifdef ASYNC_LDS
#if __has_builtin(__builtin_amdgcn_s_wait_asynccnt)
  __builtin_amdgcn_s_wait_asynccnt(N);
#else
  asm volatile("s_wait_asynccnt %0" ::"i"(N) : "memory");
#endif
#endif
}

// ---------------------------------------------------------------------------
// 1) weight convert: f32 -> f16 transposed (Bt[n][k]) for contiguous B frags
// ---------------------------------------------------------------------------
__global__ __launch_bounds__(256) void k_wconvert(const float* __restrict__ wqkv,
                                                  const float* __restrict__ wout,
                                                  half_t* __restrict__ wTqkv,
                                                  half_t* __restrict__ wTout) {
  int idx = blockIdx.x * 256 + threadIdx.x;
  if (idx < NQKV * CH) {
    int d = idx >> 9, c = idx & 511;
    wTqkv[idx] = (half_t)wqkv[c * NQKV + d];
  } else {
    int j = idx - NQKV * CH;
    int co = j >> 9, dd = j & 511;
    wTout[j] = (half_t)wout[dd * CH + co];
  }
}

// ---------------------------------------------------------------------------
// 2) GroupNorm statistics: one block per (batch, group)
// ---------------------------------------------------------------------------
__global__ __launch_bounds__(256) void k_gnstats(const float* __restrict__ x,
                                                 float2* __restrict__ stats) {
  __shared__ float ssum[256];
  __shared__ float ssq[256];
  int b = blockIdx.x >> 5;
  int g = blockIdx.x & 31;
  const float* xb = x + (size_t)b * NTOK * CH + g * 16;
  float s = 0.f, q = 0.f;
#pragma unroll 4
  for (int it = 0; it < 16; ++it) {
    int e4 = threadIdx.x + it * 256;
    int sp = e4 >> 2, j4 = (e4 & 3) * 4;
    const float4 v = *(const float4*)(xb + (size_t)sp * CH + j4);
    s += v.x + v.y + v.z + v.w;
    q += v.x * v.x + v.y * v.y + v.z * v.z + v.w * v.w;
  }
  ssum[threadIdx.x] = s;
  ssq[threadIdx.x]  = q;
  __syncthreads();
  for (int off = 128; off > 0; off >>= 1) {
    if (threadIdx.x < (unsigned)off) {
      ssum[threadIdx.x] += ssum[threadIdx.x + off];
      ssq[threadIdx.x]  += ssq[threadIdx.x + off];
    }
    __syncthreads();
  }
  if (threadIdx.x == 0) {
    float mean = ssum[0] * (1.f / 16384.f);
    float var  = ssq[0] * (1.f / 16384.f) - mean * mean;
    stats[blockIdx.x] = make_float2(mean, rsqrtf(var + EPS));
  }
}

// ---------------------------------------------------------------------------
// 3) normalize + SiLU + cast f16
// ---------------------------------------------------------------------------
__global__ __launch_bounds__(256) void k_normsilu(const float* __restrict__ x,
                                                  const float* __restrict__ scale,
                                                  const float* __restrict__ offset,
                                                  const float2* __restrict__ stats,
                                                  half_t* __restrict__ xn) {
  int idx = blockIdx.x * 256 + threadIdx.x;
  int c4 = idx & 127, row = idx >> 7;
  int b = row >> 10;
  int c = c4 * 4, g = c >> 4;
  float2 st = stats[b * 32 + g];
  const float4 xv = *(const float4*)(x + (size_t)row * CH + c);
  const float4 sc = *(const float4*)(scale + c);
  const float4 of = *(const float4*)(offset + c);
  float xs[4] = {xv.x, xv.y, xv.z, xv.w};
  float ss[4] = {sc.x, sc.y, sc.z, sc.w};
  float os[4] = {of.x, of.y, of.z, of.w};
  v4h o;
#pragma unroll
  for (int i = 0; i < 4; ++i) {
    float v = (xs[i] - st.x) * st.y * ss[i] + os[i];
    o[i] = (half_t)(v / (1.f + __expf(-v)));
  }
  *(v4h*)(xn + (size_t)row * CH + c) = o;
}

// ---------------------------------------------------------------------------
// 4) QKV GEMM (software pipelined): wave = 32 rows x 64 cols (8 C tiles)
//    block = 8 waves = 256 rows x 64 cols; grid = 64 x 24 = 1536
// ---------------------------------------------------------------------------
__global__ __launch_bounds__(256) void k_qkv_gemm(const half_t* __restrict__ xn,
                                                  const half_t* __restrict__ wT,
                                                  half_t* __restrict__ qw,
                                                  half_t* __restrict__ kw,
                                                  half_t* __restrict__ vw) {
  int lane = threadIdx.x & 31, wave = threadIdx.x >> 5;
  int nb = blockIdx.x % 24, mb = blockIdx.x / 24;
  int mrow0 = mb * 256 + wave * 32;
  int ncol0 = nb * 64;
  int lhi = lane >> 4, lm = lane & 15, kbs = lhi * 8;
  const half_t* arow[2];
  arow[0] = xn + (size_t)(mrow0 + lm) * CH;
  arow[1] = xn + (size_t)(mrow0 + 16 + lm) * CH;
  const half_t* brow[4];
#pragma unroll
  for (int t = 0; t < 4; ++t) brow[t] = wT + (size_t)(ncol0 + t * 16 + lm) * CH;

  v8f acc[2][4] = {};
  v16h a0[2], b0[4];
#pragma unroll
  for (int m = 0; m < 2; ++m) a0[m] = frag_from(arow[m], 0, kbs);
#pragma unroll
  for (int t = 0; t < 4; ++t) b0[t] = frag_from(brow[t], 0, kbs);

#pragma unroll
  for (int k0 = 0; k0 < CH; k0 += 32) {
    v16h a1[2], b1[4];
    if (k0 + 32 < CH) {            // folds at compile time (full unroll)
#pragma unroll
      for (int m = 0; m < 2; ++m) a1[m] = frag_from(arow[m], k0 + 32, kbs);
#pragma unroll
      for (int t = 0; t < 4; ++t) b1[t] = frag_from(brow[t], k0 + 32, kbs);
    }
#pragma unroll
    for (int t = 0; t < 4; ++t) {
      acc[0][t] = wmma_f16(a0[0], b0[t], acc[0][t]);
      acc[1][t] = wmma_f16(a0[1], b0[t], acc[1][t]);
    }
#pragma unroll
    for (int m = 0; m < 2; ++m) a0[m] = a1[m];
#pragma unroll
    for (int t = 0; t < 4; ++t) b0[t] = b1[t];
  }

#pragma unroll
  for (int mi = 0; mi < 2; ++mi) {
#pragma unroll
    for (int t = 0; t < 4; ++t) {
#pragma unroll
      for (int r = 0; r < 8; ++r) {
        float v = acc[mi][t][r];
        int grow = mrow0 + mi * 16 + r + 8 * lhi;
        int d = ncol0 + t * 16 + lm;
        int bb = grow >> 10, n = grow & 1023;
        int head = (d & 511) >> 6, dch = d & 63;
        int bh = bb * HEADS + head;
        if (d < 512) {
          qw[((size_t)bh * NTOK + n) * HEAD_CH + dch] = (half_t)(v * 0.125f);
        } else if (d < 1024) {
          kw[((size_t)bh * NTOK + n) * HEAD_CH + dch] = (half_t)v;
        } else {
          vw[((size_t)bh * HEAD_CH + dch) * NTOK + n] = (half_t)v;  // transposed
        }
      }
    }
  }
}

// ---------------------------------------------------------------------------
// 5) flash attention, async double-buffered K/V chunks of 64 columns
//    block per (b, head, 128-row q block); 8 waves x 16 q rows
// ---------------------------------------------------------------------------
__global__ __launch_bounds__(256) void k_attn(const half_t* __restrict__ qw,
                                              const half_t* __restrict__ kw,
                                              const half_t* __restrict__ vw,
                                              half_t* __restrict__ ow) {
  __shared__ __align__(16) half_t Kch[2][64 * 64];   // [j][d]
  __shared__ __align__(16) half_t Vch[2][64 * 64];   // [d][j]
  __shared__ __align__(16) half_t Pbuf[8 * 16 * 32]; // per-wave P scratch
  int lane = threadIdx.x & 31, wave = threadIdx.x >> 5;
  int lhi = lane >> 4, lm = lane & 15, kbs = lhi * 8;
  int bh = blockIdx.x >> 3, qblk = blockIdx.x & 7;
  int qrow0 = qblk * 128 + wave * 16;

  const half_t* qrow = qw + ((size_t)bh * NTOK + qrow0 + lm) * HEAD_CH;
  v16h aQ0 = frag_from(qrow, 0, kbs);
  v16h aQ1 = frag_from(qrow, 32, kbs);
  v8f accO[4] = {};
  float rowM[8], rowL[8];
#pragma unroll
  for (int r = 0; r < 8; ++r) { rowM[r] = -__builtin_inff(); rowL[r] = 0.f; }
  half_t* Pw = Pbuf + wave * (16 * 32);
  const half_t* kbase = kw + (size_t)bh * NTOK * HEAD_CH;
  const half_t* vbase = vw + (size_t)bh * HEAD_CH * NTOK;

  // per-chunk cooperative stage: 512 x 16B pieces each for K and V
  // (2 K + 2 V async instructions per thread -> 4 per wave per chunk)
  auto load_chunk = [&](int chn, int bufi) {
    int col0 = chn * 64;
#pragma unroll
    for (int it = 0; it < 2; ++it) {
      int idx = threadIdx.x + it * 256;
      int kr = idx >> 3, kp = (idx & 7) * 8;
      cp16(&Kch[bufi][kr * 64 + kp],
           kbase + (size_t)(col0 + kr) * HEAD_CH + kp);
      cp16(&Vch[bufi][kr * 64 + kp],
           vbase + (size_t)kr * NTOK + col0 + kp);
    }
  };

  load_chunk(0, 0);
  for (int chn = 0; chn < 16; ++chn) {
    int bufi = chn & 1;
    if (chn + 1 < 16) {
      load_chunk(chn + 1, bufi ^ 1);
      wait_async<4>();   // in-order: chunk chn done, next chunk in flight
    } else {
      wait_async<0>();
    }
    __syncthreads();

    const half_t* Kc = &Kch[bufi][0];
    const half_t* Vc = &Vch[bufi][0];
#pragma unroll
    for (int j0 = 0; j0 < 64; j0 += 32) {
      const half_t* kr0 = Kc + (j0 + lm) * 64;
      const half_t* kr1 = Kc + (j0 + 16 + lm) * 64;
      v8f s0 = {}, s1 = {};
      s0 = wmma_f16(aQ0, frag_from(kr0, 0, kbs), s0);
      s0 = wmma_f16(aQ1, frag_from(kr0, 32, kbs), s0);
      s1 = wmma_f16(aQ0, frag_from(kr1, 0, kbs), s1);
      s1 = wmma_f16(aQ1, frag_from(kr1, 32, kbs), s1);

#pragma unroll
      for (int r = 0; r < 8; ++r) {
        float mx = redmax16(fmaxf(s0[r], s1[r]));
        float mn = fmaxf(rowM[r], mx);
        float alpha = __expf(rowM[r] - mn);
        rowM[r] = mn;
        rowL[r] *= alpha;
#pragma unroll
        for (int t = 0; t < 4; ++t) accO[t][r] *= alpha;
        float p0 = __expf(s0[r] - mn);
        float p1 = __expf(s1[r] - mn);
        rowL[r] += redsum16(p0 + p1);
        int m = r + 8 * lhi;
        Pw[m * 32 + lm]      = (half_t)p0;   // C-layout -> LDS
        Pw[m * 32 + 16 + lm] = (half_t)p1;
      }
      asm volatile("s_wait_dscnt 0" ::: "memory");   // intra-wave LDS RAW
      v16h aP = frag_from(Pw + lm * 32, 0, kbs);     // reload in A-layout
#pragma unroll
      for (int t = 0; t < 4; ++t) {
        const half_t* vrow = Vc + (t * 16 + lm) * 64;
        accO[t] = wmma_f16(aP, frag_from(vrow, j0, kbs), accO[t]);
      }
    }
    __syncthreads();
  }

#pragma unroll
  for (int t = 0; t < 4; ++t) {
#pragma unroll
    for (int r = 0; r < 8; ++r) {
      int m = r + 8 * lhi;
      int grow = (bh >> 3) * NTOK + qrow0 + m;
      int gcol = (bh & 7) * HEAD_CH + t * 16 + lm;
      ow[(size_t)grow * HIDDEN + gcol] = (half_t)(accO[t][r] / rowL[r]);
    }
  }
}

// ---------------------------------------------------------------------------
// 6) out projection + bias + residual (pipelined like k_qkv_gemm)
//    grid = 64 x 8 = 512 blocks
// ---------------------------------------------------------------------------
__global__ __launch_bounds__(256) void k_outproj(const half_t* __restrict__ att,
                                                 const half_t* __restrict__ wT,
                                                 const float* __restrict__ bias,
                                                 const float* __restrict__ x,
                                                 float* __restrict__ out) {
  int lane = threadIdx.x & 31, wave = threadIdx.x >> 5;
  int nb = blockIdx.x & 7, mb = blockIdx.x >> 3;
  int mrow0 = mb * 256 + wave * 32;
  int ncol0 = nb * 64;
  int lhi = lane >> 4, lm = lane & 15, kbs = lhi * 8;
  const half_t* arow[2];
  arow[0] = att + (size_t)(mrow0 + lm) * HIDDEN;
  arow[1] = att + (size_t)(mrow0 + 16 + lm) * HIDDEN;
  const half_t* brow[4];
#pragma unroll
  for (int t = 0; t < 4; ++t) brow[t] = wT + (size_t)(ncol0 + t * 16 + lm) * HIDDEN;

  v8f acc[2][4] = {};
  v16h a0[2], b0[4];
#pragma unroll
  for (int m = 0; m < 2; ++m) a0[m] = frag_from(arow[m], 0, kbs);
#pragma unroll
  for (int t = 0; t < 4; ++t) b0[t] = frag_from(brow[t], 0, kbs);

#pragma unroll
  for (int k0 = 0; k0 < HIDDEN; k0 += 32) {
    v16h a1[2], b1[4];
    if (k0 + 32 < HIDDEN) {
#pragma unroll
      for (int m = 0; m < 2; ++m) a1[m] = frag_from(arow[m], k0 + 32, kbs);
#pragma unroll
      for (int t = 0; t < 4; ++t) b1[t] = frag_from(brow[t], k0 + 32, kbs);
    }
#pragma unroll
    for (int t = 0; t < 4; ++t) {
      acc[0][t] = wmma_f16(a0[0], b0[t], acc[0][t]);
      acc[1][t] = wmma_f16(a0[1], b0[t], acc[1][t]);
    }
#pragma unroll
    for (int m = 0; m < 2; ++m) a0[m] = a1[m];
#pragma unroll
    for (int t = 0; t < 4; ++t) b0[t] = b1[t];
  }

#pragma unroll
  for (int mi = 0; mi < 2; ++mi) {
#pragma unroll
    for (int t = 0; t < 4; ++t) {
#pragma unroll
      for (int r = 0; r < 8; ++r) {
        int grow = mrow0 + mi * 16 + r + 8 * lhi;
        int gcol = ncol0 + t * 16 + lm;
        size_t gi = (size_t)grow * CH + gcol;
        out[gi] = acc[mi][t][r] + bias[gcol] + x[gi];
      }
    }
  }
}

// ---------------------------------------------------------------------------
extern "C" void kernel_launch(void* const* d_in, const int* in_sizes, int n_in,
                              void* d_out, int out_size, void* d_ws, size_t ws_size,
                              hipStream_t stream) {
  const float* x         = (const float*)d_in[0];
  const float* gn_scale  = (const float*)d_in[1];
  const float* gn_offset = (const float*)d_in[2];
  const float* w_qkv     = (const float*)d_in[3];
  const float* w_out     = (const float*)d_in[4];
  const float* b_out     = (const float*)d_in[5];
  float* out = (float*)d_out;
  (void)in_sizes; (void)n_in; (void)out_size; (void)ws_size;

  char* ws = (char*)d_ws;
  size_t off = 0;
  auto alloc = [&](size_t bytes) -> void* {
    void* p = ws + off;
    off += (bytes + 255) & ~(size_t)255;
    return p;
  };
  float2* stats = (float2*)alloc((size_t)BATCH * GROUPS * sizeof(float2));
  half_t* wTqkv = (half_t*)alloc((size_t)NQKV * CH * 2);
  half_t* wTout = (half_t*)alloc((size_t)CH * HIDDEN * 2);
  half_t* xn    = (half_t*)alloc((size_t)ROWS * CH * 2);   // reused as attn out
  half_t* qw    = (half_t*)alloc((size_t)ROWS * CH * 2);
  half_t* kw    = (half_t*)alloc((size_t)ROWS * CH * 2);
  half_t* vw    = (half_t*)alloc((size_t)ROWS * CH * 2);

  k_wconvert<<<4096, 256, 0, stream>>>(w_qkv, w_out, wTqkv, wTout);
  k_gnstats<<<BATCH * GROUPS, 256, 0, stream>>>(x, stats);
  k_normsilu<<<8192, 256, 0, stream>>>(x, gn_scale, gn_offset, stats, xn);
  k_qkv_gemm<<<1536, 256, 0, stream>>>(xn, wTqkv, qw, kw, vw);
  k_attn<<<BATCH * HEADS * 8, 256, 0, stream>>>(qw, kw, vw, xn);
  k_outproj<<<512, 256, 0, stream>>>(xn, wTout, b_out, x, out);
}
